// Model_4174708211865
// MI455X (gfx1250) — matrix operand order repaired and checked by reading, not compiled, not run on hardware
//
#include <hip/hip_runtime.h>
#include <hip/hip_bf16.h>

// ---------------------------------------------------------------------------
// CDNA5 (gfx1250) VQ-VAE forward.
//  - All convs via implicit-GEMM on v_wmma_f32_16x16x32_bf16 (wave32).
//  - Fast path: 128-thread block = 4 waves x 16 rows, sharing a 16*NT-column
//    weight chunk staged in LDS (ds_load_b128) with register prefetch of the
//    next chunk; activations are bf16 NHWC -> contiguous B128 A-fragments.
//  - Generic gather path only for enc_c1 (Cin=3).
// ---------------------------------------------------------------------------

typedef __attribute__((ext_vector_type(16))) __bf16         v16bf;
typedef __attribute__((ext_vector_type(8)))  float          v8f;
typedef __attribute__((ext_vector_type(8)))  unsigned short u16x8;

union ABuf {
    v16bf v;
    u16x8 h[2];
    unsigned short s[16];
};

__device__ inline unsigned short f2bf(float f) {
    unsigned int u = __float_as_uint(f);
    u += 0x7FFFu + ((u >> 16) & 1u);
    return (unsigned short)(u >> 16);
}

// ---------------------------------------------------------------------------
// Fast conv. K ordered as (kh*KW+kw)*Cin + ci ; Cin multiple of 32.
// Block: 128 threads = 4 waves; wave w handles rows m0+16w..m0+16w+15,
// all waves share cols n0..n0+16*NT-1 whose B chunk lives in LDS.
//   MODE 0: ih = oh*stride - pad + kh
//   MODE 1 (transposed): ih = (oh + pad - kh)/stride when divisible
// ---------------------------------------------------------------------------
template <int NT, int MODE>
__global__ __launch_bounds__(128)
void conv_wmma_fast(const unsigned short* __restrict__ act,   // bf16 NHWC
                    const unsigned short* __restrict__ Bp,    // packed bf16
                    const float* __restrict__ bias,           // or null
                    const float* __restrict__ res,            // f32 NHWC or null
                    float* __restrict__ outF,                 // f32 out or null
                    unsigned short* __restrict__ outB,        // bf16 NHWC out or null
                    int Cin, int H, int W, int Cout, int Coutp,
                    int KH, int KW, int stride, int pad,
                    int OH, int OW, int reluF, int reluB, int outNCHW)
{
    __shared__ unsigned short smem[NT * 512];   // one B chunk: 16*NT cols x 32 k

    const int  tid  = threadIdx.x;
    const int  wave = tid >> 5;
    const int  lane = tid & 31;
    const int  lh   = lane & 15;
    const bool hiL  = lane >= 16;
    const int  OHW  = OH * OW;
    const int  m0   = (blockIdx.x * 4 + wave) * 16;
    const int  n0   = blockIdx.y * 16 * NT;

    // Per-lane A row (M always a multiple of 64 here).
    const int mA  = m0 + lh;
    const int nA  = mA / OHW;
    const int rA  = mA - nA * OHW;
    const int ohA = rA / OW;
    const int owA = rA - ohA * OW;
    const size_t actBatch = (size_t)nA * H * W * Cin;
    const int aoff = hiL ? 2 : 0;     // u16x8 index offset inside 32-elem chunk
    const int eo   = hiL ? 16 : 0;    // element offset inside B chunk / LDS

    const int cpt = Cin >> 5;         // chunks per tap
    const int nch = KH * KW * cpt;    // total K chunks (== global chunk index)
    const int nv  = NT * 64;          // u16x8 vectors per B chunk

    v8f zeroAcc = {};
    v8f acc[NT];
#pragma unroll
    for (int t = 0; t < NT; ++t) acc[t] = zeroAcc;

    u16x8* smv = (u16x8*)smem;
    u16x8 stg[2];

    // prefetch regs <- B chunk 0
    {
        const u16x8* src = (const u16x8*)(Bp + ((size_t)0 * Coutp + n0) * 32);
#pragma unroll 2
        for (int i = tid, j = 0; i < nv; i += 128, ++j) stg[j] = src[i];
    }

    int tap = 0, cidx = 0, khc = 0, kwc = 0;
    const u16x8* ap = nullptr;
    bool ok = false;

    for (int c = 0; c < nch; ++c) {
        if (cidx == 0) {
            int ih, iw;
            if (MODE == 0) {
                ih = ohA * stride - pad + khc;
                iw = owA * stride - pad + kwc;
                ok = (ih >= 0) & (ih < H) & (iw >= 0) & (iw < W);
            } else {
                int th = ohA + pad - khc;
                int tw = owA + pad - kwc;
                ih = th / stride;
                iw = tw / stride;
                ok = (th >= 0) & (tw >= 0) & (th - ih * stride == 0)
                   & (tw - iw * stride == 0) & (ih < H) & (iw < W);
            }
            ap = (const u16x8*)(act + actBatch + ((size_t)ih * W + iw) * Cin);
            if (ok) __builtin_prefetch((const void*)ap, 0, 0);
        }

        // publish staged B chunk c to LDS (all waves), then prefetch chunk c+1
        __syncthreads();
#pragma unroll 2
        for (int i = tid, j = 0; i < nv; i += 128, ++j) smv[i] = stg[j];
        __syncthreads();
        if (c + 1 < nch) {
            const u16x8* src = (const u16x8*)(Bp + ((size_t)(c + 1) * Coutp + n0) * 32);
#pragma unroll 2
            for (int i = tid, j = 0; i < nv; i += 128, ++j) stg[j] = src[i];
        }

        // A fragment: lanes lo take ci {0..15}, hi take {16..31}
        const int cb = cidx << 5;
        ABuf a;
        if (ok) {
            a.h[0] = ap[(cb >> 3) + aoff];
            a.h[1] = ap[(cb >> 3) + aoff + 1];
        } else {
#pragma unroll
            for (int i2 = 0; i2 < 16; ++i2) a.s[i2] = 0;
        }

#pragma unroll
        for (int t = 0; t < NT; ++t) {
            ABuf b;
            const u16x8* lp = (const u16x8*)(smem + ((t * 16 + lh) * 32) + eo);
            b.h[0] = lp[0];
            b.h[1] = lp[1];
            acc[t] = __builtin_amdgcn_wmma_f32_16x16x32_bf16(
                false, a.v, false, b.v, (short)0, acc[t], false, false);
        }

        if (++cidx == cpt) {
            cidx = 0;
            ++tap;
            if (++kwc == KW) { kwc = 0; ++khc; }
        }
    }

    // Epilogue. C layout: lanes 0-15 col=lane rows v..v+7; lanes 16-31 rows v+8..15.
#pragma unroll
    for (int t = 0; t < NT; ++t) {
        int col = n0 + t * 16 + lh;
        if (col >= Cout) continue;
        float bsv = (bias != nullptr) ? bias[col] : 0.0f;
#pragma unroll
        for (int v = 0; v < 8; ++v) {
            int mrow = m0 + v + (hiL ? 8 : 0);
            float val = acc[t][v] + bsv;
            size_t nidx = (size_t)mrow * Cout + col;   // NHWC flat index
            if (res != nullptr) val += res[nidx];
            if (outF != nullptr) {
                float vf = reluF ? fmaxf(val, 0.0f) : val;
                if (outNCHW) {
                    int n  = mrow / OHW;
                    int rr = mrow - n * OHW;
                    int oh = rr / OW;
                    int ow = rr - oh * OW;
                    outF[(((size_t)n * Cout + col) * OH + oh) * OW + ow] = vf;
                } else {
                    outF[nidx] = vf;
                }
            }
            if (outB != nullptr) {
                float vb = reluB ? fmaxf(val, 0.0f) : val;
                outB[nidx] = f2bf(vb);
            }
        }
    }
}

// ---------------------------------------------------------------------------
// Weight pre-pack: Bp[(chunk*Coutp + col)*32 + kk] = bf16(w[...]),
// k = chunk*32 + kk = tap*Cin + ci.
//   mode 0: w[Cout][Cin][KH][KW] ; mode 1: w[Cin][Cout][KH][KW]
// ---------------------------------------------------------------------------
__global__ void pack_w(const float* __restrict__ w, unsigned short* __restrict__ bp,
                       int K, int Cin, int KK, int Cout, int Coutp, int mode)
{
    int idx = blockIdx.x * 256 + threadIdx.x;
    int total = (K >> 5) * Coutp * 32;
    if (idx >= total) return;
    int kk    = idx & 31;
    int rest  = idx >> 5;
    int col   = rest % Coutp;
    int chunk = rest / Coutp;
    int k     = chunk * 32 + kk;
    float v = 0.0f;
    if (col < Cout) {
        int tap = k / Cin;
        int ci  = k - tap * Cin;
        v = (mode == 0) ? w[((size_t)col * Cin + ci) * KK + tap]
                        : w[((size_t)ci * Cout + col) * KK + tap];
    }
    bp[idx] = f2bf(v);
}

// ---------------------------------------------------------------------------
// Generic gather conv (only enc_c1, Cin=3). Output f32 NCHW.
// ---------------------------------------------------------------------------
__global__ __launch_bounds__(32)
void conv_wmma_generic(const float* __restrict__ in, const float* __restrict__ w,
                       const float* __restrict__ bias, float* __restrict__ out,
                       int Bn, int Cin, int H, int W, int Cout,
                       int KH, int KW, int stride, int pad,
                       int OH, int OW, int outRelu)
{
    const int  lane = threadIdx.x;
    const int  lh   = lane & 15;
    const bool hiL  = lane >= 16;
    const int  OHW  = OH * OW;
    const int  M    = Bn * OHW;
    const int  KK   = KH * KW;
    const int  K    = Cin * KK;
    const int  m0   = blockIdx.x * 16;
    const int  n0   = blockIdx.y * 16;

    const int mA  = m0 + lh;
    const int nA  = mA / OHW;
    const int rA  = mA - nA * OHW;
    const int ohA = rA / OW;
    const int owA = rA - ohA * OW;
    const float* inBase = in + (size_t)nA * Cin * H * W;
    const int colB = n0 + lh;

    v8f acc = {};
    for (int kb = 0; kb < K; kb += 32) {
        ABuf a, b;
#pragma unroll
        for (int e = 0; e < 16; ++e) {
            int ka = kb + (hiL ? 8 : 0) + ((e & 8) ? 16 : 0) + (e & 7);
            float av = 0.0f;
            if (ka < K) {
                int ci = ka / KK, r2 = ka - ci * KK;
                int kh = r2 / KW, kw = r2 - kh * KW;
                int ih = ohA * stride - pad + kh;
                int iw = owA * stride - pad + kw;
                if ((ih >= 0) & (ih < H) & (iw >= 0) & (iw < W))
                    av = inBase[((size_t)ci * H + ih) * W + iw];
            }
            a.s[e] = f2bf(av);

            int kbi = kb + (hiL ? 8 : 0) + ((e & 8) ? 16 : 0) + (e & 7);
            float bv = 0.0f;
            if (kbi < K && colB < Cout) {
                int ci = kbi / KK, r2 = kbi - ci * KK;
                bv = w[((size_t)colB * Cin + ci) * KK + r2];
            }
            b.s[e] = f2bf(bv);
        }
        acc = __builtin_amdgcn_wmma_f32_16x16x32_bf16(
                  false, a.v, false, b.v, (short)0, acc, false, false);
    }

    float bsv = (colB < Cout) ? bias[colB] : 0.0f;
#pragma unroll
    for (int v = 0; v < 8; ++v) {
        int mrow = m0 + v + (hiL ? 8 : 0);
        if (mrow < M && colB < Cout) {
            int n = mrow / OHW, rr = mrow - n * OHW;
            int oh = rr / OW, ow = rr - oh * OW;
            float val = acc[v] + bsv;
            if (outRelu) val = fmaxf(val, 0.0f);
            out[(((size_t)n * Cout + colB) * OH + oh) * OW + ow] = val;
        }
    }
}

// NCHW f32 -> NHWC bf16
__global__ void nchw_to_nhwc_bf16(const float* __restrict__ in,
                                  unsigned short* __restrict__ out,
                                  int C, int HW, int total)
{
    int i = blockIdx.x * 256 + threadIdx.x;
    if (i >= total) return;
    int p = i % HW;
    int c = (i / HW) % C;
    int n = i / (HW * C);
    out[((size_t)n * HW + p) * C + c] = f2bf(in[i]);
}

// ---------------------------------------------------------------------------
// VQ: z f32 NHWC [rows][64]; exact fp32 argmin over 512 codes; quantized
// written directly as bf16 NHWC (decoder consumes bf16 anyway).
// ---------------------------------------------------------------------------
__global__ __launch_bounds__(256)
void vq_kernel(const float* __restrict__ z, const float* __restrict__ emb,
               unsigned short* __restrict__ qb, int* __restrict__ hist,
               double* __restrict__ lossAcc, int Nrows, int Kcodes)
{
    int r = blockIdx.x * blockDim.x + threadIdx.x;
    if (r >= Nrows) return;
    const float* zp = z + (size_t)r * 64;

    float zr[64];
#pragma unroll
    for (int d = 0; d < 64; ++d) zr[d] = zp[d];

    int best = 0;
    float bestd = 3.4e38f;
    for (int j = 0; j < Kcodes; ++j) {
        const float* e = emb + (size_t)j * 64;   // wave-uniform -> cache broadcast
        float dist = 0.0f;
#pragma unroll
        for (int d = 0; d < 64; ++d) {
            float df = zr[d] - e[d];
            dist = fmaf(df, df, dist);
        }
        if (dist < bestd) { bestd = dist; best = j; }
    }

    const float* e = emb + (size_t)best * 64;
    unsigned short* qp = qb + (size_t)r * 64;
    float err = 0.0f;
#pragma unroll
    for (int d = 0; d < 64; ++d) {
        float ev = e[d];
        qp[d] = f2bf(ev);
        float df = ev - zr[d];
        err = fmaf(df, df, err);
    }
    atomicAdd(lossAcc, (double)err);
    atomicAdd(&hist[best], 1);
}

__global__ void zero_kernel(double* lossAcc, int* hist)
{
    int i = blockIdx.x * blockDim.x + threadIdx.x;
    if (i == 0) *lossAcc = 0.0;
    if (i < 512) hist[i] = 0;
}

__global__ void finalize_kernel(const double* lossAcc, const int* __restrict__ hist,
                                float* outLoss, float* outPerp,
                                float invN, float lossScale)
{
    if (threadIdx.x == 0 && blockIdx.x == 0) {
        *outLoss = (float)(lossAcc[0]) * lossScale;   // (1+0.25)*mean((q-z)^2)
        float s = 0.0f;
        for (int k = 0; k < 512; ++k) {
            float p = (float)hist[k] * invN;
            s += p * logf(p + 1e-10f);
        }
        *outPerp = expf(-s);
    }
}

extern "C" void kernel_launch(void* const* d_in, const int* in_sizes, int n_in,
                              void* d_out, int out_size, void* d_ws, size_t ws_size,
                              hipStream_t stream)
{
    (void)in_sizes; (void)n_in; (void)out_size; (void)ws_size;

    const float* x    = (const float*)d_in[0];
    const float* ec1w = (const float*)d_in[1];
    const float* ec1b = (const float*)d_in[2];
    const float* ec2w = (const float*)d_in[3];
    const float* ec2b = (const float*)d_in[4];
    const float* ec3w = (const float*)d_in[5];
    const float* ec3b = (const float*)d_in[6];
    const float* erw1 = (const float*)d_in[7];   // [2,32,128,3,3]
    const float* erw2 = (const float*)d_in[8];   // [2,128,32,1,1]
    const float* prew = (const float*)d_in[9];
    const float* preb = (const float*)d_in[10];
    const float* emb  = (const float*)d_in[11];  // [512,64]
    const float* dc1w = (const float*)d_in[12];
    const float* dc1b = (const float*)d_in[13];
    const float* drw1 = (const float*)d_in[14];
    const float* drw2 = (const float*)d_in[15];
    const float* dt1w = (const float*)d_in[16];  // [128,64,4,4] (in,out,k,k)
    const float* dt1b = (const float*)d_in[17];
    const float* dt2w = (const float*)d_in[18];  // [64,3,4,4]
    const float* dt2b = (const float*)d_in[19];

    float* out     = (float*)d_out;
    float* outLoss = out;
    float* xrec    = out + 1;
    float* outPerp = out + 1 + (size_t)64 * 3 * 128 * 128;

    float*  ws      = (float*)d_ws;
    double* lossAcc = (double*)ws;
    int*    hist    = (int*)(ws + 2);
    size_t off = 1024;
    auto FBUF = [&](size_t nFloats) { float* p = ws + off; off += nFloats; return p; };
    auto HBUF = [&](size_t nElems)  { unsigned short* p = (unsigned short*)(ws + off);
                                      off += nElems / 2; return p; };

    float*          z1n = FBUF((size_t)64 * 64 * 64 * 64);   // enc_c1 out f32 NCHW
    unsigned short* z1b = HBUF((size_t)64 * 64 * 64 * 64);   // bf16 NHWC
    unsigned short* z2b = HBUF((size_t)64 * 32 * 32 * 128);
    float*          z3f = FBUF((size_t)64 * 32 * 32 * 128);  // residual trunk f32 NHWC
    unsigned short* z3b = HBUF((size_t)64 * 32 * 32 * 128);  // relu'd bf16
    unsigned short* hbb = HBUF((size_t)64 * 32 * 32 * 32);
    float*          zqf = FBUF((size_t)64 * 32 * 32 * 64);   // z_e f32 NHWC
    unsigned short* qzb = HBUF((size_t)64 * 32 * 32 * 64);   // quantized bf16 NHWC
    float*          d1f = FBUF((size_t)64 * 32 * 32 * 128);
    unsigned short* d1b = HBUF((size_t)64 * 32 * 32 * 128);
    unsigned short* t1b = HBUF((size_t)64 * 64 * 64 * 64);

    // packed weights
    unsigned short* pkEc2  = HBUF(1024 * 128);
    unsigned short* pkEc3  = HBUF(1152 * 128);
    unsigned short* pkEr1a = HBUF(1152 * 32);
    unsigned short* pkEr1b = HBUF(1152 * 32);
    unsigned short* pkEr2a = HBUF(32 * 128);
    unsigned short* pkEr2b = HBUF(32 * 128);
    unsigned short* pkPre  = HBUF(128 * 64);
    unsigned short* pkDc1  = HBUF(576 * 128);
    unsigned short* pkDr1a = HBUF(1152 * 32);
    unsigned short* pkDr1b = HBUF(1152 * 32);
    unsigned short* pkDr2a = HBUF(32 * 128);
    unsigned short* pkDr2b = HBUF(32 * 128);
    unsigned short* pkDt1  = HBUF(2048 * 64);
    unsigned short* pkDt2  = HBUF(1024 * 16);

    auto coutp = [](int Cout, int NT) { int g = 16 * NT; return ((Cout + g - 1) / g) * g; };

    auto pack = [&](const float* w, unsigned short* bp, int Cin, int KK, int Cout,
                    int NT, int mode) {
        int K = Cin * KK;
        int Cp = coutp(Cout, NT);
        int total = (K >> 5) * Cp * 32;
        pack_w<<<dim3((total + 255) / 256), dim3(256), 0, stream>>>(
            w, bp, K, Cin, KK, Cout, Cp, mode);
    };

    auto fast = [&](const unsigned short* act, const unsigned short* bp,
                    const float* bias, const float* res, float* outF,
                    unsigned short* outB, int Bn, int Cin, int H, int W,
                    int Cout, int NT, int KH, int KW, int st, int pd,
                    int OH, int OW, int mode, int reluF, int reluB, int nchw) {
        int M  = Bn * OH * OW;
        int Cp = coutp(Cout, NT);
        dim3 g(M / 64, Cp / (16 * NT));
        dim3 b(128);
        if (NT == 4 && mode == 0)
            conv_wmma_fast<4, 0><<<g, b, 0, stream>>>(act, bp, bias, res, outF, outB,
                Cin, H, W, Cout, Cp, KH, KW, st, pd, OH, OW, reluF, reluB, nchw);
        else if (NT == 2 && mode == 0)
            conv_wmma_fast<2, 0><<<g, b, 0, stream>>>(act, bp, bias, res, outF, outB,
                Cin, H, W, Cout, Cp, KH, KW, st, pd, OH, OW, reluF, reluB, nchw);
        else if (NT == 4 && mode == 1)
            conv_wmma_fast<4, 1><<<g, b, 0, stream>>>(act, bp, bias, res, outF, outB,
                Cin, H, W, Cout, Cp, KH, KW, st, pd, OH, OW, reluF, reluB, nchw);
        else
            conv_wmma_fast<1, 1><<<g, b, 0, stream>>>(act, bp, bias, res, outF, outB,
                Cin, H, W, Cout, Cp, KH, KW, st, pd, OH, OW, reluF, reluB, nchw);
    };

    // ---- weight packing (tiny, once per launch) ----
    pack(ec2w, pkEc2, 64, 16, 128, 4, 0);
    pack(ec3w, pkEc3, 128, 9, 128, 4, 0);
    pack(erw1,                pkEr1a, 128, 9, 32, 2, 0);
    pack(erw1 + 32 * 128 * 9, pkEr1b, 128, 9, 32, 2, 0);
    pack(erw2,                pkEr2a, 32, 1, 128, 4, 0);
    pack(erw2 + 128 * 32,     pkEr2b, 32, 1, 128, 4, 0);
    pack(prew, pkPre, 128, 1, 64, 4, 0);
    pack(dc1w, pkDc1, 64, 9, 128, 4, 0);
    pack(drw1,                pkDr1a, 128, 9, 32, 2, 0);
    pack(drw1 + 32 * 128 * 9, pkDr1b, 128, 9, 32, 2, 0);
    pack(drw2,                pkDr2a, 32, 1, 128, 4, 0);
    pack(drw2 + 128 * 32,     pkDr2b, 32, 1, 128, 4, 0);
    pack(dt1w, pkDt1, 128, 16, 64, 4, 1);
    pack(dt2w, pkDt2, 64, 16, 3, 1, 1);

    zero_kernel<<<dim3(2), dim3(256), 0, stream>>>(lossAcc, hist);

    // ---- encoder ----
    {   // relu(conv 3->64 k4 s2 p1) 128->64, generic gather path (Cin=3)
        dim3 g((64 * 64 * 64) / 16, 4);
        conv_wmma_generic<<<g, dim3(32), 0, stream>>>(x, ec1w, ec1b, z1n,
            64, 3, 128, 128, 64, 4, 4, 2, 1, 64, 64, 1);
    }
    {   int total = 64 * 64 * 64 * 64;
        nchw_to_nhwc_bf16<<<dim3((total + 255) / 256), dim3(256), 0, stream>>>(
            z1n, z1b, 64, 64 * 64, total);
    }
    // relu(conv 64->128 k4 s2 p1) 64->32
    fast(z1b, pkEc2, ec2b, nullptr, nullptr, z2b, 64, 64, 64, 64, 128, 4,
         4, 4, 2, 1, 32, 32, 0, 0, 1, 0);
    // conv 128->128 k3 p1 (f32 trunk + relu'd bf16 for res conv1)
    fast(z2b, pkEc3, ec3b, nullptr, z3f, z3b, 64, 128, 32, 32, 128, 4,
         3, 3, 1, 1, 32, 32, 0, 0, 1, 0);
    // residual stack x2: h = conv1(relu(x)); x += conv2(relu(h))
    const unsigned short* er1[2] = {pkEr1a, pkEr1b};
    const unsigned short* er2[2] = {pkEr2a, pkEr2b};
    for (int i = 0; i < 2; ++i) {
        fast(z3b, er1[i], nullptr, nullptr, nullptr, hbb, 64, 128, 32, 32, 32, 2,
             3, 3, 1, 1, 32, 32, 0, 0, 1, 0);
        fast(hbb, er2[i], nullptr, z3f, z3f, z3b, 64, 32, 32, 32, 128, 4,
             1, 1, 1, 0, 32, 32, 0, 0, 1, 0);
    }
    // pre-VQ conv 128->64 k1 on relu(trunk) -> f32 NHWC
    fast(z3b, pkPre, preb, nullptr, zqf, nullptr, 64, 128, 32, 32, 64, 4,
         1, 1, 1, 0, 32, 32, 0, 0, 0, 0);

    // ---- vector quantizer ----
    vq_kernel<<<dim3(256), dim3(256), 0, stream>>>(zqf, emb, qzb, hist, lossAcc,
                                                   64 * 32 * 32, 512);

    // ---- decoder ----
    fast(qzb, pkDc1, dc1b, nullptr, d1f, d1b, 64, 64, 32, 32, 128, 4,
         3, 3, 1, 1, 32, 32, 0, 0, 1, 0);
    const unsigned short* dr1[2] = {pkDr1a, pkDr1b};
    const unsigned short* dr2[2] = {pkDr2a, pkDr2b};
    for (int i = 0; i < 2; ++i) {
        fast(d1b, dr1[i], nullptr, nullptr, nullptr, hbb, 64, 128, 32, 32, 32, 2,
             3, 3, 1, 1, 32, 32, 0, 0, 1, 0);
        fast(hbb, dr2[i], nullptr, d1f, d1f, d1b, 64, 32, 32, 32, 128, 4,
             1, 1, 1, 0, 32, 32, 0, 0, 1, 0);
    }
    // relu(convT 128->64 k4 s2 p1) 32->64 (input already relu'd via d1b)
    fast(d1b, pkDt1, dt1b, nullptr, nullptr, t1b, 64, 128, 32, 32, 64, 4,
         4, 4, 2, 1, 64, 64, 1, 0, 1, 0);
    // convT 64->3 k4 s2 p1 64->128, straight to d_out as f32 NCHW
    fast(t1b, pkDt2, dt2b, nullptr, xrec, nullptr, 64, 64, 64, 64, 3, 1,
         4, 4, 2, 1, 128, 128, 1, 0, 0, 1);

    finalize_kernel<<<dim3(1), dim3(1), 0, stream>>>(lossAcc, hist, outLoss, outPerp,
                                                     1.0f / 65536.0f,
                                                     1.25f / 4194304.0f);
}
